// Siamese_37941741093598
// MI455X (gfx1250) — matrix operand order repaired and checked by reading, not compiled, not run on hardware
//
#include <hip/hip_runtime.h>
#include <hip/hip_bf16.h>

// ---------------------------------------------------------------------------
// Siamese LSTM (S=512, B=256, E=H=256, V=50000) for gfx1250 / MI455X.
// Recurrence-bound: batch is the only parallel axis -> 32 persistent
// workgroups (2 sequences x 16 batch-tiles of M=16), each running the full
// 512-step recurrence with v_wmma_f32_16x16x32_bf16 (f32 accumulate).
// Weights (bf16, 1 MB total) stream from L2 every step.  An opaque zero
// offset (asm volatile "+s") keeps the loads inside the t-loop (no spill);
// per-K-step batches of 8 named B-fragments (Wih, then Whh) let the loads issue as clause
// bursts with partial waits instead of one full wait per WMMA.
// ---------------------------------------------------------------------------

typedef __bf16 bf16;
typedef __attribute__((ext_vector_type(16))) __bf16 v16bf;
typedef __attribute__((ext_vector_type(8)))  __bf16 v8bf;
typedef __attribute__((ext_vector_type(8)))  float  v8f;

typedef const __attribute__((address_space(1))) v8bf* gv8bf;

#define SEQ   512
#define BATCH 256
#define HID   256
#define GATES 1024   // 4*H, PyTorch gate order i,f,g,o

// ---- WMMA fragment loaders -------------------------------------------------
// 16-bit A matrix 16x32 (MxK): m = lane&15 for both halves; lanes 0-15 hold
// K = k0..k0+15, lanes 16-31 K = k0+16..k0+31.  Each half is a
// contiguous 16-byte run -> one ds_load_b128 each.
__device__ inline v16bf load_a_frag(const bf16* __restrict__ row, int k0, int sel) {
    v8bf lo = *reinterpret_cast<const v8bf*>(row + k0 + sel * 16);
    v8bf hi = *reinterpret_cast<const v8bf*>(row + k0 + sel * 16 + 8);
    return __builtin_shufflevector(lo, hi, 0,1,2,3,4,5,6,7,8,9,10,11,12,13,14,15);
}

// 16-bit B matrix 32x16 (KxN): B[k,n] = W[n_global, k], weights [1024][256]
// bf16 row-major.  n = lane&15; lanes 0-15 take K [k0..k0+15], lanes 16-31
// K [k0+16..k0+31]: one 32-byte run per lane -> two global_load_b128
// (address_space(1) guarantees the global, LOADcnt-only form).
__device__ inline v16bf load_b_frag(const bf16* w, size_t idx) {
    gv8bf p = (gv8bf)(unsigned long long)(w + idx);
    v8bf lo = p[0];
    v8bf hi = p[1];
    return __builtin_shufflevector(lo, hi, 0,1,2,3,4,5,6,7,8,9,10,11,12,13,14,15);
}

__device__ inline float sigmoidf_(float x) { return 1.0f / (1.0f + __expf(-x)); }

// ---- prep: fp32 -> bf16 weights, fused bias --------------------------------
__global__ __launch_bounds__(256)
void prep_kernel(const float* __restrict__ wih_f, const float* __restrict__ whh_f,
                 const float* __restrict__ b_ih,  const float* __restrict__ b_hh,
                 bf16* __restrict__ wih, bf16* __restrict__ whh, float* __restrict__ bias) {
    int i = blockIdx.x * blockDim.x + threadIdx.x;
    const int n = GATES * HID;
    for (int idx = i; idx < n; idx += gridDim.x * blockDim.x) {
        wih[idx] = (bf16)wih_f[idx];
        whh[idx] = (bf16)whh_f[idx];
    }
    if (i < GATES) bias[i] = b_ih[i] + b_hh[i];
}

// ---- persistent LSTM recurrence -------------------------------------------
// grid = 32 blocks (seq 0/1 x 16 batch tiles), 256 threads = 8 wave32.
// Wave w owns gate columns [w*128, w*128+128) -> 8 WMMA N-tiles; each wave's
// columns lie entirely inside one gate (i/f/g/o) since gates split at 256.
// Thread tid permanently owns hidden column j = tid: cell state c[16] and
// its 4 bias scalars stay in registers for all 512 steps.
__global__ __launch_bounds__(256, 1)
void lstm_kernel(const int* __restrict__ s1, const int* __restrict__ s2,
                 const float* __restrict__ emb,
                 const float* __restrict__ h0_1, const float* __restrict__ c0_1,
                 const float* __restrict__ h0_2, const float* __restrict__ c0_2,
                 const bf16* __restrict__ wih, const bf16* __restrict__ whh,
                 const float* __restrict__ bias, float* __restrict__ feat) {
    __shared__ bf16  xa[16][HID];      // embedded inputs, bf16 row-major (8 KB)
    __shared__ bf16  ha[16][HID];      // hidden state,    bf16 row-major (8 KB)
    __shared__ float gsm[16][GATES];   // raw gate pre-activations        (64 KB)

    const int g    = blockIdx.x;
    const int seq  = g >> 4;           // 0 -> s1, 1 -> s2
    const int b0   = (g & 15) * 16;    // batch-tile base row
    const int* toks      = seq ? s2   : s1;
    const float* h0      = seq ? h0_2 : h0_1;
    const float* c0      = seq ? c0_2 : c0_1;

    const int tid  = threadIdx.x;
    const int lane = tid & 31;
    const int wave = tid >> 5;
    const int m16  = lane & 15;        // M (or N) index within a 16x16 tile
    const int sel  = lane >> 4;        // which K half this lane covers

    // ---- init: h0 -> LDS bf16, c0 + per-column biases -> registers ----
    float creg[16];
#pragma unroll
    for (int b = 0; b < 16; ++b) {
        creg[b]    = c0[(size_t)(b0 + b) * HID + tid];
        ha[b][tid] = (bf16)h0[(size_t)(b0 + b) * HID + tid];
    }
    const float bias_i = bias[tid];
    const float bias_f = bias[tid + 256];
    const float bias_g = bias[tid + 512];
    const float bias_o = bias[tid + 768];

    const v8f vzero = {};    // loop-invariant zero C-operand (materialized once)
    unsigned wofs = 0;       // opaque zero: defeats LICM of the weight loads

    for (int t = 0; t < SEQ; ++t) {
        // ---- gather embeddings for this step into xa (bf16) ----
        {
            const int r  = tid >> 4;          // batch row 0..15
            const int ck = (tid & 15) * 16;   // 16-column chunk
            const int tok = toks[t * BATCH + b0 + r];
            const float4* src = reinterpret_cast<const float4*>(emb + (size_t)tok * HID + ck);
#pragma unroll
            for (int q = 0; q < 4; ++q) {
                float4 v = src[q];
                xa[r][ck + q * 4 + 0] = (bf16)v.x;
                xa[r][ck + q * 4 + 1] = (bf16)v.y;
                xa[r][ck + q * 4 + 2] = (bf16)v.z;
                xa[r][ck + q * 4 + 3] = (bf16)v.w;
            }
            // hide next step's random-access embedding gather behind the GEMM
            if (t + 1 < SEQ) {
                const int tok2 = toks[(t + 1) * BATCH + b0 + r];
                __builtin_prefetch(emb + (size_t)tok2 * HID + ck, 0, 0);
            }
        }
        __syncthreads();   // xa + ha ready; prior step's gsm consumers done

        // launder the (always-zero) offset so weight addresses are
        // loop-variant: loads stay inside the t-loop, no spill, no volatile.
        asm volatile("" : "+s"(wofs));
        const bf16* wih_t = wih + wofs;
        const bf16* whh_t = whh + wofs;

        // ---- gates = x @ Wih^T + h @ Whh^T  (bf16 WMMA, f32 acc) ----
        v8f acc[8];
#pragma unroll
        for (int kk = 0; kk < 8; ++kk) {
            const int k0 = kk * 32;
            const v16bf ax = load_a_frag(&xa[m16][0], k0, sel);
            const v16bf ah = load_a_frag(&ha[m16][0], k0, sel);
            // batch all 8 Wih fragments -> clause burst, partial waits
            v16bf bx[8];
#pragma unroll
            for (int nt = 0; nt < 8; ++nt)
                bx[nt] = load_b_frag(wih_t,
                    (size_t)((wave * 8 + nt) * 16 + m16) * HID + k0 + sel * 16);
#pragma unroll
            for (int nt = 0; nt < 8; ++nt)
                acc[nt] = __builtin_amdgcn_wmma_f32_16x16x32_bf16(
                              false, ax, false, bx[nt], (short)0,
                              (kk == 0) ? vzero : acc[nt], false, false);
            // batch all 8 Whh fragments
            v16bf bh[8];
#pragma unroll
            for (int nt = 0; nt < 8; ++nt)
                bh[nt] = load_b_frag(whh_t,
                    (size_t)((wave * 8 + nt) * 16 + m16) * HID + k0 + sel * 16);
#pragma unroll
            for (int nt = 0; nt < 8; ++nt)
                acc[nt] = __builtin_amdgcn_wmma_f32_16x16x32_bf16(
                              false, ah, false, bh[nt], (short)0, acc[nt], false, false);
        }

        // ---- spill accumulators (C layout: VGPR r -> M=r+8*sel, N=lane&15) ----
#pragma unroll
        for (int nt = 0; nt < 8; ++nt) {
            const int n = (wave * 8 + nt) * 16 + m16;
#pragma unroll
            for (int r = 0; r < 8; ++r)
                gsm[r + sel * 8][n] = acc[nt][r];
        }
        __syncthreads();   // gsm ready for elementwise consumers

        // ---- elementwise cell update; thread tid owns column j = tid ----
        {
            const int j = tid;
#pragma unroll 4
            for (int b = 0; b < 16; ++b) {
                const float ig = gsm[b][j]       + bias_i;
                const float fg = gsm[b][j + 256] + bias_f;
                const float gg = gsm[b][j + 512] + bias_g;
                const float og = gsm[b][j + 768] + bias_o;
                const float i_ = sigmoidf_(ig);
                const float f_ = sigmoidf_(fg);
                const float g_ = tanhf(gg);
                const float o_ = sigmoidf_(og);
                const float c  = f_ * creg[b] + i_ * g_;
                creg[b] = c;
                const float h  = o_ * tanhf(c);
                ha[b][j] = (bf16)h;
                if (t == SEQ - 1)
                    feat[(size_t)(b0 + b) * (2 * HID) + seq * HID + j] = h;
            }
        }
        // no barrier here: next iteration's pre-GEMM barrier orders
        // ha/gsm against their next readers; the gather only writes xa.
    }
}

// ---- head: out = (feat @ W1^T + b1) @ W2^T + b2 ---------------------------
__global__ __launch_bounds__(256)
void head_kernel(const float* __restrict__ feat,
                 const float* __restrict__ W1, const float* __restrict__ b1,
                 const float* __restrict__ W2, const float* __restrict__ b2,
                 float* __restrict__ out) {
    __shared__ float p0[256];
    __shared__ float p1[256];
    const int b = blockIdx.x;     // batch row 0..255
    const int m = threadIdx.x;    // hidden unit 0..255
    const float* fr = feat + (size_t)b * (2 * HID);
    const float* wr = W1 + (size_t)m * (2 * HID);
    float acc = b1[m];
#pragma unroll 8
    for (int k = 0; k < 2 * HID; ++k) acc += fr[k] * wr[k];
    p0[m] = acc * W2[m];          // W2 row 0
    p1[m] = acc * W2[HID + m];    // W2 row 1
    __syncthreads();
    if (m < 2) {
        const float* p = m ? p1 : p0;
        float s = b2[m];
        for (int k = 0; k < 256; ++k) s += p[k];
        out[b * 2 + m] = s;
    }
}

// ---------------------------------------------------------------------------
extern "C" void kernel_launch(void* const* d_in, const int* in_sizes, int n_in,
                              void* d_out, int out_size, void* d_ws, size_t ws_size,
                              hipStream_t stream) {
    (void)in_sizes; (void)n_in; (void)out_size; (void)ws_size;
    const int*   s1   = (const int*)  d_in[0];
    const int*   s2   = (const int*)  d_in[1];
    const float* emb  = (const float*)d_in[2];
    const float* W_ih = (const float*)d_in[3];
    const float* W_hh = (const float*)d_in[4];
    const float* b_ih = (const float*)d_in[5];
    const float* b_hh = (const float*)d_in[6];
    const float* h0_1 = (const float*)d_in[7];
    const float* c0_1 = (const float*)d_in[8];
    const float* h0_2 = (const float*)d_in[9];
    const float* c0_2 = (const float*)d_in[10];
    const float* W1   = (const float*)d_in[11];
    const float* b1   = (const float*)d_in[12];
    const float* W2   = (const float*)d_in[13];
    const float* b2   = (const float*)d_in[14];
    float* out = (float*)d_out;

    // workspace layout (bytes): bf16 Wih | bf16 Whh | bias f32 | feat f32
    char* ws = (char*)d_ws;
    bf16*  wih_b = (bf16*)(ws);                       // 1024*256*2 = 512 KB
    bf16*  whh_b = (bf16*)(ws + 524288);              // 512 KB
    float* bias  = (float*)(ws + 1048576);            // 4 KB
    float* feat  = (float*)(ws + 1048576 + 4096);     // 256*512*4 = 512 KB

    prep_kernel<<<1024, 256, 0, stream>>>(W_ih, W_hh, b_ih, b_hh, wih_b, whh_b, bias);
    lstm_kernel<<<32, 256, 0, stream>>>(s1, s2, emb, h0_1, c0_1, h0_2, c0_2,
                                        wih_b, whh_b, bias, feat);
    head_kernel<<<256, 256, 0, stream>>>(feat, W1, b1, W2, b2, out);
}
